// LinearAttention_40707700031914
// MI455X (gfx1250) — matrix operand order repaired and checked by reading, not compiled, not run on hardware
//
#include <hip/hip_runtime.h>
#include <math.h>
#include <stdint.h>

typedef __attribute__((ext_vector_type(16))) __bf16 v16bf;
typedef __attribute__((ext_vector_type(8)))  float  v8f;

#define B_     8
#define C_     512
#define HWD_   96
#define HW_    9216          // 96*96
#define INNER_ 256
#define QKV_   768
#define NP_    (B_*HW_)      // 73728
#define DH_    32
#define EPS_   1e-5f
#define SCALE_ 0.17677669529663687f   // 32^-0.5

__device__ __forceinline__ unsigned short f2bf(float f) {
  __bf16 h = (__bf16)f;
  return __builtin_bit_cast(unsigned short, h);
}

// gfx1250 async global->LDS copy (tracked on ASYNCcnt). VDST VGPR holds the
// LDS byte address (low 32 bits of the generic pointer = LDS offset).
__device__ __forceinline__ void async_b128(void* lds, const void* g) {
  asm volatile("global_load_async_to_lds_b128 %0, %1, off"
               :: "v"((uint32_t)(uintptr_t)lds), "v"(g) : "memory");
}
__device__ __forceinline__ void wait_async0() {
  asm volatile("s_wait_asynccnt 0x0" ::: "memory");
}

// ---------------- 0. one-time weight f32 -> bf16 conversion -----------------
// wqkv: [768][512] (q|k|v rows), wout: [512][256]
__global__ __launch_bounds__(256) void la_wcvt(
    const float* __restrict__ wq, const float* __restrict__ wk,
    const float* __restrict__ wv, const float* __restrict__ wo,
    unsigned short* __restrict__ wqkv, unsigned short* __restrict__ wout) {
  int i = blockIdx.x * 256 + threadIdx.x;      // 0 .. 524287
  if (i < 393216) {
    float v;
    if (i < 131072)      v = wq[i];
    else if (i < 262144) v = wk[i - 131072];
    else                 v = wv[i - 262144];
    wqkv[i] = f2bf(v);
  } else {
    int j = i - 393216;                        // 0 .. 131071
    wout[j] = f2bf(wo[j]);
  }
}

// ---------------- 1. ChanLayerNorm statistics (per position over 512 ch) ----
__global__ __launch_bounds__(256) void la_ln_stats(const float* __restrict__ x,
                                                   float* __restrict__ mean,
                                                   float* __restrict__ rstd) {
  int b  = blockIdx.x;
  int hw = blockIdx.y * 256 + threadIdx.x;
  const float* xp = x + (size_t)b * C_ * HW_ + hw;
  float s = 0.f, s2 = 0.f;
  for (int c = 0; c < C_; ++c) { float v = xp[(size_t)c * HW_]; s += v; s2 += v * v; }
  float m   = s * (1.0f / C_);
  float var = s2 * (1.0f / C_) - m * m;
  int p = b * HW_ + hw;
  mean[p] = m;
  rstd[p] = rsqrtf(var + EPS_);
}

// ---------------- 2. Fused LN + q/k/v 1x1 conv GEMM (bf16 WMMA) -------------
// M = 73728 positions, N = 768 (q|k|v), K = 512. Block tile 128M x 64N.
__global__ __launch_bounds__(256) void la_qkv_gemm(
    const float* __restrict__ x, const float* __restrict__ mean,
    const float* __restrict__ rstd, const float* __restrict__ g,
    const unsigned short* __restrict__ wqkv, float* __restrict__ out) {
  __shared__ __bf16 As[128][32];
  __shared__ __bf16 Bs[64][32];
  int t = threadIdx.x, lane = t & 31, wave = t >> 5;
  int pos0 = blockIdx.x * 128;            // never crosses batch: 9216 % 128 == 0
  int b = pos0 / HW_, hw0 = pos0 % HW_;
  int n0 = blockIdx.y * 64;
  v8f acc0 = {}, acc1 = {}, acc2 = {}, acc3 = {};

  for (int k0 = 0; k0 < C_; k0 += 32) {
    __syncthreads();
    // stage B tile via async copy: 64 rows x 32 bf16 (one b128 per thread)
    {
      int nl = t >> 2, kq = t & 3;
      const unsigned short* src = wqkv + (size_t)(n0 + nl) * C_ + k0 + kq * 8;
      async_b128(&Bs[nl][kq * 8], src);
    }
    // stage A tile (apply LayerNorm on the fly), coalesced over hw
    #pragma unroll
    for (int it = 0; it < 16; ++it) {
      int idx = it * 256 + t;
      int cc = idx >> 7;                  // channel 0..31
      int m  = idx & 127;                 // position 0..127
      int p  = pos0 + m;
      float v = x[((size_t)b * C_ + (k0 + cc)) * HW_ + hw0 + m];
      v = (v - mean[p]) * rstd[p] * g[k0 + cc];
      As[m][cc] = (__bf16)v;
    }
    wait_async0();
    __syncthreads();

    // A fragment: 16x32 bf16 per ISA layout
    v16bf a;
    {
      int m = lane & 15, kb = (lane >> 4) * 8;
      #pragma unroll
      for (int i = 0; i < 16; ++i) {
        int k = kb + (i & 7) + ((i >> 3) * 16);
        a[i] = As[wave * 16 + m][k];
      }
    }
    // B fragments: 32x16 bf16 per ISA layout
    int n = lane & 15, kb2 = (lane >> 4) * 8;
    v16bf b0, b1, b2, b3;
    #pragma unroll
    for (int i = 0; i < 16; ++i) {
      int k = kb2 + (i & 7) + ((i >> 3) * 16);
      b0[i] = Bs[ 0 + n][k];
      b1[i] = Bs[16 + n][k];
      b2[i] = Bs[32 + n][k];
      b3[i] = Bs[48 + n][k];
    }
    acc0 = __builtin_amdgcn_wmma_f32_16x16x32_bf16(false, a, false, b0, (short)0, acc0, false, false);
    acc1 = __builtin_amdgcn_wmma_f32_16x16x32_bf16(false, a, false, b1, (short)0, acc1, false, false);
    acc2 = __builtin_amdgcn_wmma_f32_16x16x32_bf16(false, a, false, b2, (short)0, acc2, false, false);
    acc3 = __builtin_amdgcn_wmma_f32_16x16x32_bf16(false, a, false, b3, (short)0, acc3, false, false);
  }

  // store D to [b][768][hw]
  int n = lane & 15, mofs = (lane >> 4) * 8;
  #pragma unroll
  for (int vv = 0; vv < 8; ++vv) {
    int hw = hw0 + wave * 16 + vv + mofs;
    size_t base = (size_t)b * QKV_ * HW_ + hw;
    out[base + (size_t)(n0 + n     ) * HW_] = acc0[vv];
    out[base + (size_t)(n0 + n + 16) * HW_] = acc1[vv];
    out[base + (size_t)(n0 + n + 32) * HW_] = acc2[vv];
    out[base + (size_t)(n0 + n + 48) * HW_] = acc3[vv];
  }
}

// ---------------- 3. Depthwise 3x3 SAME for all 768 channels ----------------
__global__ __launch_bounds__(256) void la_dwconv(const float* __restrict__ in,
    const float* __restrict__ wq, const float* __restrict__ wk,
    const float* __restrict__ wv, float* __restrict__ out) {
  int bc = blockIdx.x;                       // b*768 + c
  int hw = blockIdx.y * 256 + threadIdx.x;
  int c = bc % QKV_;
  const float* w;
  if (c < 256)      { w = wq + (size_t)c * 9; }
  else if (c < 512) { w = wk + (size_t)(c - 256) * 9; }
  else              { w = wv + (size_t)(c - 512) * 9; }
  int y = hw / HWD_, xq = hw % HWD_;
  const float* ip = in + (size_t)bc * HW_;
  float s = 0.f;
  #pragma unroll
  for (int dy = -1; dy <= 1; ++dy)
    #pragma unroll
    for (int dx = -1; dx <= 1; ++dx) {
      int yy = y + dy, xx = xq + dx;
      if (yy >= 0 && yy < HWD_ && xx >= 0 && xx < HWD_)
        s += w[(dy + 1) * 3 + (dx + 1)] * ip[yy * HWD_ + xx];
    }
  out[(size_t)bc * HW_ + hw] = s;
}

// ---------------- 4. q softmax over feature dim (dh=32) * scale, in place ---
__global__ __launch_bounds__(256) void la_q_softmax(float* __restrict__ q) {
  int bh = blockIdx.x, b = bh >> 3, h = bh & 7;
  int n = blockIdx.y * 256 + threadIdx.x;
  float* base = q + ((size_t)b * QKV_ + h * DH_) * HW_ + n;
  float v[32];
  float mx = -1e30f;
  #pragma unroll
  for (int d = 0; d < 32; ++d) { v[d] = base[(size_t)d * HW_]; mx = fmaxf(mx, v[d]); }
  float s = 0.f;
  #pragma unroll
  for (int d = 0; d < 32; ++d) { v[d] = __expf(v[d] - mx); s += v[d]; }
  float inv = SCALE_ / s;
  #pragma unroll
  for (int d = 0; d < 32; ++d) base[(size_t)d * HW_] = v[d] * inv;
}

// ---------------- 5. k softmax over spatial dim (9216), in place ------------
__global__ __launch_bounds__(256) void la_k_softmax(float* __restrict__ kbuf) {
  int b = blockIdx.x >> 8, ci = blockIdx.x & 255;
  float* row = kbuf + ((size_t)b * QKV_ + 256 + ci) * HW_;
  __shared__ float red[256];
  int t = threadIdx.x;
  float mx = -1e30f;
  for (int n = t; n < HW_; n += 256) mx = fmaxf(mx, row[n]);
  red[t] = mx; __syncthreads();
  for (int s = 128; s > 0; s >>= 1) { if (t < s) red[t] = fmaxf(red[t], red[t + s]); __syncthreads(); }
  mx = red[0]; __syncthreads();
  float sum = 0.f;
  for (int n = t; n < HW_; n += 256) sum += __expf(row[n] - mx);
  red[t] = sum; __syncthreads();
  for (int s = 128; s > 0; s >>= 1) { if (t < s) red[t] += red[t + s]; __syncthreads(); }
  float inv = 1.0f / red[0];
  for (int n = t; n < HW_; n += 256) row[n] = __expf(row[n] - mx) * inv;
}

// ---------------- 6. context[bh][d][e] = sum_n k[n,d]*v[n,e] ----------------
__global__ __launch_bounds__(1024) void la_context(const float* __restrict__ kv,
                                                   float* __restrict__ ctx) {
  int bh = blockIdx.x, b = bh >> 3, h = bh & 7;
  int d = threadIdx.x >> 5, e = threadIdx.x & 31;
  const float4* k4 = (const float4*)(kv + ((size_t)b * QKV_ + 256 + h * DH_ + d) * HW_);
  const float4* v4 = (const float4*)(kv + ((size_t)b * QKV_ + 512 + h * DH_ + e) * HW_);
  float acc = 0.f;
  for (int n = 0; n < HW_ / 4; ++n) {
    float4 a = k4[n], c = v4[n];
    acc += a.x * c.x + a.y * c.y + a.z * c.z + a.w * c.w;
  }
  ctx[(size_t)bh * 1024 + d * 32 + e] = acc;
}

// ---------------- 7. out = q @ ctx, SiLU, -> bf16 [Np][256] -----------------
__global__ __launch_bounds__(256) void la_attn_silu(const float* __restrict__ qkv2,
                                                    const float* __restrict__ ctx,
                                                    unsigned short* __restrict__ attn) {
  __shared__ float cs[32][32];
  int bh = blockIdx.x, b = bh >> 3, h = bh & 7;
  int t = threadIdx.x;
  #pragma unroll
  for (int i = 0; i < 4; ++i) {
    int idx = i * 256 + t;
    cs[idx >> 5][idx & 31] = ctx[(size_t)bh * 1024 + idx];
  }
  __syncthreads();
  int n = blockIdx.y * 256 + t;
  const float* qb = qkv2 + ((size_t)b * QKV_ + h * DH_) * HW_ + n;
  float qv[32];
  #pragma unroll
  for (int d = 0; d < 32; ++d) qv[d] = qb[(size_t)d * HW_];
  size_t op = ((size_t)(b * HW_ + n)) * INNER_ + h * DH_;
  #pragma unroll
  for (int e = 0; e < 32; ++e) {
    float s = 0.f;
    #pragma unroll
    for (int d = 0; d < 32; ++d) s += qv[d] * cs[d][e];
    float sig = 1.0f / (1.0f + __expf(-s));
    attn[op + e] = f2bf(s * sig);
  }
}

// ---------------- 8. out 1x1 conv GEMM (bf16 WMMA) -> d_out [b][512][hw] ----
// A and B tiles both staged with async global->LDS b128 copies.
__global__ __launch_bounds__(256) void la_out_gemm(const unsigned short* __restrict__ A,
                                                   const unsigned short* __restrict__ w,
                                                   float* __restrict__ out) {
  __shared__ __bf16 As[128][32];
  __shared__ __bf16 Bs[64][32];
  int t = threadIdx.x, lane = t & 31, wave = t >> 5;
  int pos0 = blockIdx.x * 128;
  int b = pos0 / HW_, hw0 = pos0 % HW_;
  int n0 = blockIdx.y * 64;
  v8f acc0 = {}, acc1 = {}, acc2 = {}, acc3 = {};

  for (int k0 = 0; k0 < INNER_; k0 += 32) {
    __syncthreads();
    // A tile: 128 rows x 32 bf16 = 512 x 16B chunks, 2 per thread
    #pragma unroll
    for (int it = 0; it < 2; ++it) {
      int chunk = it * 256 + t;
      int m = chunk >> 2, q = chunk & 3;
      async_b128(&As[m][q * 8], A + (size_t)(pos0 + m) * INNER_ + k0 + q * 8);
    }
    // B tile: 64 rows x 32 bf16, one 16B chunk per thread
    {
      int nl = t >> 2, kq = t & 3;
      async_b128(&Bs[nl][kq * 8], w + (size_t)(n0 + nl) * INNER_ + k0 + kq * 8);
    }
    wait_async0();
    __syncthreads();

    v16bf a;
    {
      int m = lane & 15, kb = (lane >> 4) * 8;
      #pragma unroll
      for (int i = 0; i < 16; ++i) {
        int k = kb + (i & 7) + ((i >> 3) * 16);
        a[i] = As[wave * 16 + m][k];
      }
    }
    int n = lane & 15, kb2 = (lane >> 4) * 8;
    v16bf b0, b1, b2, b3;
    #pragma unroll
    for (int i = 0; i < 16; ++i) {
      int k = kb2 + (i & 7) + ((i >> 3) * 16);
      b0[i] = Bs[ 0 + n][k];
      b1[i] = Bs[16 + n][k];
      b2[i] = Bs[32 + n][k];
      b3[i] = Bs[48 + n][k];
    }
    acc0 = __builtin_amdgcn_wmma_f32_16x16x32_bf16(false, a, false, b0, (short)0, acc0, false, false);
    acc1 = __builtin_amdgcn_wmma_f32_16x16x32_bf16(false, a, false, b1, (short)0, acc1, false, false);
    acc2 = __builtin_amdgcn_wmma_f32_16x16x32_bf16(false, a, false, b2, (short)0, acc2, false, false);
    acc3 = __builtin_amdgcn_wmma_f32_16x16x32_bf16(false, a, false, b3, (short)0, acc3, false, false);
  }

  int n = lane & 15, mofs = (lane >> 4) * 8;
  #pragma unroll
  for (int vv = 0; vv < 8; ++vv) {
    int hw = hw0 + wave * 16 + vv + mofs;
    size_t base = (size_t)b * C_ * HW_ + hw;
    out[base + (size_t)(n0 + n     ) * HW_] = acc0[vv];
    out[base + (size_t)(n0 + n + 16) * HW_] = acc1[vv];
    out[base + (size_t)(n0 + n + 32) * HW_] = acc2[vv];
    out[base + (size_t)(n0 + n + 48) * HW_] = acc3[vv];
  }
}

// ---------------- 9. final ChanLayerNorm, in place on d_out -----------------
__global__ __launch_bounds__(256) void la_final_ln(float* __restrict__ y,
                                                   const float* __restrict__ g) {
  int b  = blockIdx.x;
  int hw = blockIdx.y * 256 + threadIdx.x;
  float* yp = y + (size_t)b * C_ * HW_ + hw;
  float s = 0.f, s2 = 0.f;
  for (int c = 0; c < C_; ++c) { float v = yp[(size_t)c * HW_]; s += v; s2 += v * v; }
  float m   = s * (1.0f / C_);
  float var = s2 * (1.0f / C_) - m * m;
  float r = rsqrtf(var + EPS_);
  for (int c = 0; c < C_; ++c) {
    float v = yp[(size_t)c * HW_];
    yp[(size_t)c * HW_] = (v - m) * r * g[c];
  }
}

extern "C" void kernel_launch(void* const* d_in, const int* in_sizes, int n_in,
                              void* d_out, int out_size, void* d_ws, size_t ws_size,
                              hipStream_t stream) {
  (void)in_sizes; (void)n_in; (void)out_size; (void)ws_size;
  const float* fmap   = (const float*)d_in[0];
  const float* norm_g = (const float*)d_in[1];
  const float* q_w1   = (const float*)d_in[2];
  const float* q_wd   = (const float*)d_in[3];
  const float* k_w1   = (const float*)d_in[4];
  const float* k_wd   = (const float*)d_in[5];
  const float* v_w1   = (const float*)d_in[6];
  const float* v_wd   = (const float*)d_in[7];
  const float* out_w  = (const float*)d_in[8];
  const float* out_ng = (const float*)d_in[9];
  float* out = (float*)d_out;

  char* ws = (char*)d_ws;
  size_t o = 0;
  float* mean = (float*)(ws + o); o += (size_t)NP_ * 4;
  float* rstd = (float*)(ws + o); o += (size_t)NP_ * 4;
  float* qkv1 = (float*)(ws + o); o += (size_t)B_ * QKV_ * HW_ * 4;
  float* qkv2 = (float*)(ws + o); o += (size_t)B_ * QKV_ * HW_ * 4;
  float* ctx  = (float*)(ws + o); o += (size_t)64 * 1024 * 4;
  unsigned short* attn = (unsigned short*)(ws + o); o += (size_t)NP_ * INNER_ * 2;
  unsigned short* wqkv = (unsigned short*)(ws + o); o += (size_t)QKV_ * C_ * 2;
  unsigned short* wout = (unsigned short*)(ws + o);

  la_wcvt     <<<dim3(2048), 256, 0, stream>>>(q_w1, k_w1, v_w1, out_w, wqkv, wout);
  la_ln_stats <<<dim3(B_, HW_ / 256), 256, 0, stream>>>(fmap, mean, rstd);
  la_qkv_gemm <<<dim3(NP_ / 128, QKV_ / 64), 256, 0, stream>>>(
      fmap, mean, rstd, norm_g, wqkv, qkv1);
  la_dwconv   <<<dim3(B_ * QKV_, HW_ / 256), 256, 0, stream>>>(
      qkv1, q_wd, k_wd, v_wd, qkv2);
  la_q_softmax<<<dim3(64, HW_ / 256), 256, 0, stream>>>(qkv2);
  la_k_softmax<<<dim3(B_ * 256), 256, 0, stream>>>(qkv2);
  la_context  <<<dim3(64), 1024, 0, stream>>>(qkv2, ctx);
  la_attn_silu<<<dim3(64, HW_ / 256), 256, 0, stream>>>(qkv2, ctx, attn);
  la_out_gemm <<<dim3(NP_ / 128, C_ / 64), 256, 0, stream>>>(attn, wout, out);
  la_final_ln <<<dim3(B_, HW_ / 256), 256, 0, stream>>>(out, out_ng);
}
